// MultiheadSelfAttention_48790828482832
// MI455X (gfx1250) — matrix operand order repaired and checked by reading, not compiled, not run on hardware
//
#include <hip/hip_runtime.h>

// Sizes fixed by the reference
#define EMB   1024
#define NH    16
#define HD    64
#define SEQ   2048
#define BATCH 2
#define MROWS (BATCH * SEQ)     // 4096

typedef __attribute__((ext_vector_type(16))) __bf16 v16bf;
typedef __attribute__((ext_vector_type(8)))  __bf16 v8bf;
typedef __attribute__((ext_vector_type(8)))  float  v8f;
typedef __attribute__((ext_vector_type(4)))  unsigned su4;
typedef __attribute__((ext_vector_type(8)))  unsigned su8;

// ---------------------------------------------------------------------------
// WMMA fragment loaders (CDNA5 16x16x32 bf16 layouts, CDNA5 ISA)
// ---------------------------------------------------------------------------
__device__ __forceinline__ v16bf load_afrag(const __bf16* base, int stride, int lane) {
  const int row = lane & 15;
  const int kb  = (lane >> 4) << 4;
  const __bf16* p = base + row * stride + kb;
  v8bf lo = *(const v8bf*)(p);
  v8bf hi = *(const v8bf*)(p + 8);
  return __builtin_shufflevector(lo, hi, 0,1,2,3,4,5,6,7,8,9,10,11,12,13,14,15);
}

__device__ __forceinline__ v16bf load_bfrag(const __bf16* base, int stride, int lane) {
  const int n  = lane & 15;
  const int kb = (lane >> 4) << 4;
  const __bf16* p = base + n * stride + kb;
  v8bf lo = *(const v8bf*)(p);
  v8bf hi = *(const v8bf*)(p + 8);
  return __builtin_shufflevector(lo, hi, 0,1,2,3,4,5,6,7,8,9,10,11,12,13,14,15);
}

// B fragment gathered from a K-major tile [K][N] with row pitch (elems).
__device__ __forceinline__ v16bf load_bfrag_kmajor(const __bf16* base, int pitch, int lane) {
  const int n  = lane & 15;
  const int kb = (lane >> 4) << 4;
  v16bf r;
#pragma unroll
  for (int e = 0; e < 16; ++e) r[e] = base[(kb + e) * pitch + n];
  return r;
}

__device__ __forceinline__ v8f wmma_bf16(v16bf a, v16bf b, v8f c) {
  return __builtin_amdgcn_wmma_f32_16x16x32_bf16(false, a, false, b, (short)0, c,
                                                 false, false);
}

// ---------------------------------------------------------------------------
// CDNA5 async global->LDS copies.  INST_OFFSET is added to BOTH the LDS and
// the global address, so one LDS VGPR + one 64-bit VGPR pair serves a whole
// contiguous run via immediate offsets.
// ---------------------------------------------------------------------------
__device__ __forceinline__ void wait_async_0()  { asm volatile("s_wait_asynccnt 0x0" ::: "memory"); }
__device__ __forceinline__ void wait_async_12() { asm volatile("s_wait_asynccnt 0xc" ::: "memory"); }

__device__ __forceinline__ unsigned lds_off_of(const void* p) {
  return (unsigned)(uintptr_t)p;   // low 32 bits of LDS aperture addr = LDS offset
}

// 12 async b128 ops: A row (64B) + two B rows (64B each), double-buffer static
__device__ __forceinline__ void gemm_issue12(unsigned aL, unsigned b0L, unsigned b1L,
                                             unsigned long long ga,
                                             unsigned long long gb0,
                                             unsigned long long gb1) {
  asm volatile(
      "global_load_async_to_lds_b128 %0, %3, off\n\t"
      "global_load_async_to_lds_b128 %0, %3, off offset:16\n\t"
      "global_load_async_to_lds_b128 %0, %3, off offset:32\n\t"
      "global_load_async_to_lds_b128 %0, %3, off offset:48\n\t"
      "global_load_async_to_lds_b128 %1, %4, off\n\t"
      "global_load_async_to_lds_b128 %1, %4, off offset:16\n\t"
      "global_load_async_to_lds_b128 %1, %4, off offset:32\n\t"
      "global_load_async_to_lds_b128 %1, %4, off offset:48\n\t"
      "global_load_async_to_lds_b128 %2, %5, off\n\t"
      "global_load_async_to_lds_b128 %2, %5, off offset:16\n\t"
      "global_load_async_to_lds_b128 %2, %5, off offset:32\n\t"
      "global_load_async_to_lds_b128 %2, %5, off offset:48"
      :: "v"(aL), "v"(b0L), "v"(b1L), "v"(ga), "v"(gb0), "v"(gb1)
      : "memory");
}

// ---------------------------------------------------------------------------
// Kernel 0a: fp32 -> bf16 flat convert
// ---------------------------------------------------------------------------
__global__ void k_cvt(const float* __restrict__ src, __bf16* __restrict__ dst, int n) {
  int i = blockIdx.x * blockDim.x + threadIdx.x;
  if (i < n) dst[i] = (__bf16)src[i];
}

// Kernel 0b: transpose + convert  W[K][N] (fp32) -> Wt[N][K] (bf16)
__global__ void k_tr(const float* __restrict__ src, __bf16* __restrict__ dst,
                     int K, int N) {
  int i = blockIdx.x * blockDim.x + threadIdx.x;
  if (i >= K * N) return;
  int k = i / N, n = i - k * N;
  dst[n * K + k] = (__bf16)src[i];
}

// ---------------------------------------------------------------------------
// Shared GEMM core: one wave computes a 32(M) x 64(N) tile of A @ B^T,
// K = EMB, async double-buffered LDS, static buffer parity (2x unroll).
// ---------------------------------------------------------------------------
__device__ __forceinline__ void gemm_compute(const __bf16* A, const __bf16* Bt,
                                             int lane, v8f acc[8]) {
  const v16bf a0 = load_afrag(A, 32, lane);
  const v16bf a1 = load_afrag(A + 512, 32, lane);
#pragma unroll
  for (int nt = 0; nt < 4; ++nt) {
    const v16bf b = load_bfrag(Bt + nt * 512, 32, lane);
    acc[nt]     = wmma_bf16(a0, b, acc[nt]);
    acc[nt + 4] = wmma_bf16(a1, b, acc[nt + 4]);
  }
}

__device__ __forceinline__ void gemm_core(const __bf16* gA, const __bf16* gB,
                                          __bf16* ldsA, __bf16* ldsB,
                                          int lane, v8f acc[8]) {
  const unsigned aoff = lds_off_of(ldsA);               // [2][32*32] bf16
  const unsigned boff = lds_off_of(ldsB);               // [2][64*32] bf16
  const unsigned aL0  = aoff + (unsigned)lane * 64;
  const unsigned aL1  = aL0 + 2048;
  const unsigned b0L0 = boff + (unsigned)lane * 64;
  const unsigned b1L0 = b0L0 + 2048;                    // rows 32..63
  const unsigned b0L1 = b0L0 + 4096;
  const unsigned b1L1 = b1L0 + 4096;
  unsigned long long ga  = (unsigned long long)(uintptr_t)(gA + (size_t)lane * EMB);
  unsigned long long gb0 = (unsigned long long)(uintptr_t)(gB + (size_t)lane * EMB);
  unsigned long long gb1 = (unsigned long long)(uintptr_t)(gB + (size_t)(lane + 32) * EMB);

  gemm_issue12(aL0, b0L0, b1L0, ga, gb0, gb1);
  ga += 64; gb0 += 64; gb1 += 64;
  for (int it = 0; it < 15; ++it) {                     // 30 steady chunks
    gemm_issue12(aL1, b0L1, b1L1, ga, gb0, gb1);
    ga += 64; gb0 += 64; gb1 += 64;
    wait_async_12();
    gemm_compute(ldsA, ldsB, lane, acc);
    gemm_issue12(aL0, b0L0, b1L0, ga, gb0, gb1);
    ga += 64; gb0 += 64; gb1 += 64;
    wait_async_12();
    gemm_compute(ldsA + 1024, ldsB + 2048, lane, acc);
  }
  gemm_issue12(aL1, b0L1, b1L1, ga, gb0, gb1);          // chunk 31
  wait_async_12();
  gemm_compute(ldsA, ldsB, lane, acc);                  // chunk 30
  wait_async_0();
  gemm_compute(ldsA + 1024, ldsB + 2048, lane, acc);    // chunk 31
}

// ---------------------------------------------------------------------------
// Kernel 1: QKV projection -> bf16 Q/K/V buffers laid out [B][H][S][D]
// ---------------------------------------------------------------------------
__global__ void __launch_bounds__(128) k_qkv(const __bf16* __restrict__ xb,
                                             const __bf16* __restrict__ wt,
                                             const float*  __restrict__ bias,
                                             __bf16* __restrict__ Qb,
                                             __bf16* __restrict__ Kb,
                                             __bf16* __restrict__ Vb) {
  __shared__ __align__(16) __bf16 smA[4][2 * 32 * 32];
  __shared__ __align__(16) __bf16 smB[4][2 * 64 * 32];
  const int wave = threadIdx.x >> 5, lane = threadIdx.x & 31;
  const int tile = blockIdx.x * 4 + wave;               // 128 x 48 tiles
  const int tm = tile / 48, tn = tile - tm * 48;
  const int m0 = tm * 32, n0 = tn * 64;

  v8f acc[8] = {};
  gemm_core(xb + (size_t)m0 * EMB, wt + (size_t)n0 * EMB,
            smA[wave], smB[wave], lane, acc);

  const int hl = (lane >> 4) << 3;
#pragma unroll
  for (int nt = 0; nt < 4; ++nt) {
    const int nn = n0 + nt * 16 + (lane & 15);
    const float bv = bias[nn];
    const int region = nn >> 10;                        // 0=Q 1=K 2=V
    const int e = nn & 1023;
    const int hh = e >> 6, dd = e & 63;
    __bf16* dst = (region == 0) ? Qb : ((region == 1) ? Kb : Vb);
#pragma unroll
    for (int r = 0; r < 8; ++r) {
      int m = m0 + r + hl;
      int bb = m >> 11, ss = m & 2047;
      dst[(((size_t)(bb * NH + hh) * SEQ) + ss) * HD + dd] = (__bf16)(acc[nt][r] + bv);
      m += 16; bb = m >> 11; ss = m & 2047;
      dst[(((size_t)(bb * NH + hh) * SEQ) + ss) * HD + dd] = (__bf16)(acc[nt + 4][r] + bv);
    }
  }
}

// ---------------------------------------------------------------------------
// Kernel 2: causal flash attention.  One wave = one (b, h, 16-row q tile).
// K chunk arrives via the Tensor Data Mover (TENSORcnt); Q and V arrive via
// async global->LDS (ASYNCcnt).  V is K-major with padded pitch 72.
// ---------------------------------------------------------------------------
#define VPITCH 72
__global__ void __launch_bounds__(128) k_attn(const __bf16* __restrict__ Qb,
                                              const __bf16* __restrict__ Kb,
                                              const __bf16* __restrict__ Vb,
                                              __bf16* __restrict__ Ob) {
  // per-wave LDS: Q 16x64 | K 32x64 | Vraw 32x72 | P 16x32 = 5888 bf16
  __shared__ __align__(16) __bf16 sm[4][5888];
  const int wave = threadIdx.x >> 5, lane = threadIdx.x & 31;
  const int wid = blockIdx.x * 4 + wave;
  const int qt = wid & 127;
  const int hh = (wid >> 7) & 15;
  const int bb = wid >> 11;
  const int qbase = qt * 16;

  __bf16* sQ = sm[wave];
  __bf16* sK = sQ + 1024;
  __bf16* sV = sK + 2048;          // [kv=32][VPITCH]
  __bf16* sP = sV + 32 * VPITCH;   // probs 16x32
  const unsigned koff = lds_off_of(sK);

  const size_t headBase = (size_t)(bb * NH + hh) * SEQ * HD;
  const __bf16* Qg  = Qb + headBase + (size_t)qbase * HD;
  const __bf16* Kg0 = Kb + headBase;
  const __bf16* Vg0 = Vb + headBase;

  // stage Q (2KB contiguous): 4 async b128 ops, one address pair
  {
    const unsigned qL = lds_off_of(sQ) + (unsigned)lane * 16;
    unsigned long long qa = (unsigned long long)(uintptr_t)Qg + (unsigned)(lane * 16);
    asm volatile(
        "global_load_async_to_lds_b128 %0, %1, off\n\t"
        "global_load_async_to_lds_b128 %0, %1, off offset:512\n\t"
        "global_load_async_to_lds_b128 %0, %1, off offset:1024\n\t"
        "global_load_async_to_lds_b128 %0, %1, off offset:1536"
        :: "v"(qL), "v"(qa) : "memory");
    wait_async_0();
  }
  const v16bf aq0 = load_afrag(sQ, 64, lane);        // d 0..31
  const v16bf aq1 = load_afrag(sQ + 32, 64, lane);   // d 32..63

  // TDM descriptor pieces that never change (group 1)
  su8 g1;
  g1[0] = 0x00010000u;                 // data_size = 1 (2 bytes/elem)
  g1[1] = (unsigned)HD << 16;          // tensor_dim0 = 64 (lo16 in [31:16])
  g1[2] = (unsigned)SEQ << 16;         // tensor_dim1 = 2048 (lo16 in [31:16])
  g1[3] = (unsigned)HD << 16;          // tile_dim0 = 64
  g1[4] = 32u;                         // tile_dim1 = 32, tile_dim2 = 0
  g1[5] = (unsigned)HD;                // tensor_dim0_stride = 64
  g1[6] = 0u;
  g1[7] = 0u;

  v8f o0 = {}, o1 = {}, o2 = {}, o3 = {};
  float mrow[8], lrow[8];
#pragma unroll
  for (int r = 0; r < 8; ++r) { mrow[r] = -3.0e38f; lrow[r] = 0.0f; }

  const int hl   = (lane >> 4) << 3;
  const int qend = qbase + 15;

  const unsigned vL = lds_off_of(sV) + (unsigned)lane * (VPITCH * 2);
  unsigned long long va = (unsigned long long)(uintptr_t)Vg0 + (unsigned)(lane * 128);
  unsigned long long ka = (unsigned long long)(uintptr_t)Kg0;

  for (int kv0 = 0; kv0 <= qend; kv0 += 32) {
    // K chunk via Tensor Data Mover: 2D tile 64x32 bf16 -> linear LDS
    {
      su4 g0;
      g0[0] = 1u;                                          // count = 1
      g0[1] = __builtin_amdgcn_readfirstlane(koff);        // lds_addr
      g0[2] = __builtin_amdgcn_readfirstlane((unsigned)ka);
      g0[3] = (__builtin_amdgcn_readfirstlane((unsigned)(ka >> 32)) & 0x01FFFFFFu)
              | 0x80000000u;                               // addr[56:32] | type=2
      asm volatile("tensor_load_to_lds %0, %1" :: "s"(g0), "s"(g1) : "memory");
    }
    // V chunk via async copies: row lane -> LDS row (pitch 144B), 8 x b128
    asm volatile(
        "global_load_async_to_lds_b128 %0, %1, off\n\t"
        "global_load_async_to_lds_b128 %0, %1, off offset:16\n\t"
        "global_load_async_to_lds_b128 %0, %1, off offset:32\n\t"
        "global_load_async_to_lds_b128 %0, %1, off offset:48\n\t"
        "global_load_async_to_lds_b128 %0, %1, off offset:64\n\t"
        "global_load_async_to_lds_b128 %0, %1, off offset:80\n\t"
        "global_load_async_to_lds_b128 %0, %1, off offset:96\n\t"
        "global_load_async_to_lds_b128 %0, %1, off offset:112"
        :: "v"(vL), "v"(va) : "memory");
    ka += 4096; va += 4096;
    wait_async_0();
    __builtin_amdgcn_s_wait_tensorcnt(0);

#pragma unroll
    for (int half = 0; half < 2; ++half) {
      const __bf16* kb = sK + half * 16 * 64;        // 16 kv rows, [N][K] form
      v8f s = {};
      s = wmma_bf16(aq0, load_bfrag(kb, 64, lane), s);
      s = wmma_bf16(aq1, load_bfrag(kb + 32, 64, lane), s);
      const int col = kv0 + half * 16 + (lane & 15);
#pragma unroll
      for (int r = 0; r < 8; ++r) {
        float sc = s[r] * 0.125f;                    // 1/sqrt(64)
        const int qr = qbase + r + hl;
        sc = (col <= qr) ? sc : -1.0e30f;            // causal mask, no branch
        float mx = sc;                               // rowmax over 16-lane half
        mx = fmaxf(mx, __shfl_xor(mx, 1));
        mx = fmaxf(mx, __shfl_xor(mx, 2));
        mx = fmaxf(mx, __shfl_xor(mx, 4));
        mx = fmaxf(mx, __shfl_xor(mx, 8));
        const float mnew = fmaxf(mrow[r], mx);
        const float corr = __expf(mrow[r] - mnew);
        const float pv   = __expf(sc - mnew);
        float sum = pv;
        sum += __shfl_xor(sum, 1);
        sum += __shfl_xor(sum, 2);
        sum += __shfl_xor(sum, 4);
        sum += __shfl_xor(sum, 8);
        lrow[r] = lrow[r] * corr + sum;
        mrow[r] = mnew;
        o0[r] *= corr; o1[r] *= corr; o2[r] *= corr; o3[r] *= corr;
        sP[(r + hl) * 32 + half * 16 + (lane & 15)] = (__bf16)pv;
      }
    }
    // O += P(16x32) @ V(32x64); V B-frags gathered K-major from sV
    const v16bf ap = load_afrag(sP, 32, lane);
    o0 = wmma_bf16(ap, load_bfrag_kmajor(sV +  0, VPITCH, lane), o0);
    o1 = wmma_bf16(ap, load_bfrag_kmajor(sV + 16, VPITCH, lane), o1);
    o2 = wmma_bf16(ap, load_bfrag_kmajor(sV + 32, VPITCH, lane), o2);
    o3 = wmma_bf16(ap, load_bfrag_kmajor(sV + 48, VPITCH, lane), o3);
  }

  // normalize + store O as [B][S][H][D] bf16 (== [M][E] row-major for proj)
#pragma unroll
  for (int r = 0; r < 8; ++r) {
    const float rl = 1.0f / lrow[r];
    const int ss = qbase + r + hl;
    const size_t rb = ((size_t)(bb * SEQ + ss) * NH + hh) * HD + (lane & 15);
    Ob[rb +  0] = (__bf16)(o0[r] * rl);
    Ob[rb + 16] = (__bf16)(o1[r] * rl);
    Ob[rb + 32] = (__bf16)(o2[r] * rl);
    Ob[rb + 48] = (__bf16)(o3[r] * rl);
  }
}

// ---------------------------------------------------------------------------
// Kernel 3: output projection.  out[4096][1024] = Ob @ WtOut^T + b_out (fp32)
// ---------------------------------------------------------------------------
__global__ void __launch_bounds__(128) k_out(const __bf16* __restrict__ Ab,
                                             const __bf16* __restrict__ wt,
                                             const float*  __restrict__ bias,
                                             float* __restrict__ out) {
  __shared__ __align__(16) __bf16 smA[4][2 * 32 * 32];
  __shared__ __align__(16) __bf16 smB[4][2 * 64 * 32];
  const int wave = threadIdx.x >> 5, lane = threadIdx.x & 31;
  const int tile = blockIdx.x * 4 + wave;               // 128 x 16 tiles
  const int tm = tile >> 4, tn = tile & 15;
  const int m0 = tm * 32, n0 = tn * 64;

  v8f acc[8] = {};
  gemm_core(Ab + (size_t)m0 * EMB, wt + (size_t)n0 * EMB,
            smA[wave], smB[wave], lane, acc);

  const int hl = (lane >> 4) << 3;
#pragma unroll
  for (int nt = 0; nt < 4; ++nt) {
    const int nn = n0 + nt * 16 + (lane & 15);
    const float bv = bias[nn];
#pragma unroll
    for (int r = 0; r < 8; ++r) {
      const int m = m0 + r + hl;
      out[(size_t)m * EMB + nn]        = acc[nt][r] + bv;
      out[(size_t)(m + 16) * EMB + nn] = acc[nt + 4][r] + bv;
    }
  }
}

// ---------------------------------------------------------------------------
// Host launcher
// ---------------------------------------------------------------------------
extern "C" void kernel_launch(void* const* d_in, const int* in_sizes, int n_in,
                              void* d_out, int out_size, void* d_ws, size_t ws_size,
                              hipStream_t stream) {
  const float* x     = (const float*)d_in[0];
  const float* W_qkv = (const float*)d_in[1];
  const float* b_qkv = (const float*)d_in[2];
  const float* W_out = (const float*)d_in[3];
  const float* b_out = (const float*)d_in[4];

  // workspace layout (bytes); total ~50.3 MB
  char* ws = (char*)d_ws;
  __bf16* xb  = (__bf16*)(ws);                       //  8,388,608  x bf16
  __bf16* wtq = (__bf16*)(ws + 8388608);             //  6,291,456  W_qkv^T
  __bf16* wto = (__bf16*)(ws + 14680064);            //  2,097,152  W_out^T
  __bf16* Qb  = (__bf16*)(ws + 16777216);            //  8,388,608  [B][H][S][D]
  __bf16* Kb  = (__bf16*)(ws + 25165824);
  __bf16* Vb  = (__bf16*)(ws + 33554432);
  __bf16* Ob  = (__bf16*)(ws + 41943040);            //  [B][S][H][D]

  const int nx = MROWS * EMB;                        // 4,194,304
  k_cvt<<<(nx + 255) / 256, 256, 0, stream>>>(x, xb, nx);
  k_tr<<<(EMB * 3 * EMB + 255) / 256, 256, 0, stream>>>(W_qkv, wtq, EMB, 3 * EMB);
  k_tr<<<(EMB * EMB + 255) / 256, 256, 0, stream>>>(W_out, wto, EMB, EMB);

  // 128 x 48 tiles of 32x64, 4 waves/block
  k_qkv<<<(128 * 48) / 4, 128, 0, stream>>>(xb, wtq, b_qkv, Qb, Kb, Vb);

  // B*H*(S/16) = 4096 waves, 4 per block
  k_attn<<<4096 / 4, 128, 0, stream>>>(Qb, Kb, Vb, Ob);

  // 128 x 16 tiles of 32x64, 4 waves/block
  k_out<<<(128 * 16) / 4, 128, 0, stream>>>(Ob, wto, b_out, (float*)d_out);
}
